// LSTMModel_28681791602980
// MI455X (gfx1250) — matrix operand order repaired and checked by reading, not compiled, not run on hardware
//
#include <hip/hip_runtime.h>

// ---------------------------------------------------------------------------
// 2-layer LSTM (B=32, T=512, H=E=1024) + embedding + FC for MI455X (gfx1250).
// Strategy: bf16 WMMA 16x16x32 with f32 accumulate. LSTM weights converted to
// bf16 once per call (32 MB, L2-resident on a 192 MB L2 part), embeddings
// gathered & packed into WMMA A-layout up front, one fused GEMM+cell kernel
// per (timestep, layer), h recurrence ping-pongs through packed bf16 buffers.
// ---------------------------------------------------------------------------

typedef __attribute__((ext_vector_type(16))) __bf16        v16bf;
typedef __attribute__((ext_vector_type(8)))  float         v8f;
typedef __attribute__((ext_vector_type(8)))  unsigned int  v8u;

#define Tdim 512
#define Bdim 32
#define Hdim 1024
#define Edim 1024
#define OUTD 32000

// round-to-nearest-even fp32 -> bf16 (bit pattern)
__device__ __forceinline__ unsigned short f2bf(float f) {
  unsigned int u = __builtin_bit_cast(unsigned int, f);
  u += 0x7FFFu + ((u >> 16) & 1u);
  return (unsigned short)(u >> 16);
}

__device__ __forceinline__ v8f wmma_bf16(v8u a, v8u b, v8f c) {
  return __builtin_amdgcn_wmma_f32_16x16x32_bf16(
      false, __builtin_bit_cast(v16bf, a),
      false, __builtin_bit_cast(v16bf, b),
      (short)0, c, false, false);
}

__device__ __forceinline__ float sigmoidf_fast(float x) {
  return 1.0f / (1.0f + __expf(-x));
}

// --- A-matrix (16x32 bf16) packed layout helpers -----------------------------
// Element (m, k): tile (mt = m/16, kt = k/32). Within tile (contiguous k order chosen here, not the ISA table order):
//   lanes 0-15 hold K in {0..15}, lanes 16-31 hold K in {16..31}
//   lane = (m%16) + 16*bit4(kk);  vgpr = (kk&15)/2;  half = kk&1
// Buffer layout: dword index = ((mt*32 + kt)*32 + lane)*8 + vgpr
__device__ __forceinline__ void a_slot(int m, int k, int& dw, int& half) {
  int mt = m >> 4, kt = k >> 5, kk = k & 31;
  int lane = (m & 15) + (((kk >> 4) & 1) << 4);
  int v = (kk & 15) >> 1;
  dw = ((mt * 32 + kt) * 32 + lane) * 8 + v;
  half = kk & 1;
}

// ---------------------------------------------------------------------------
// Embedding gather -> bf16, packed straight into WMMA A-layout [T][2mt][32kt]
// ---------------------------------------------------------------------------
__global__ void __launch_bounds__(256) embed_pack_kernel(
    const int* __restrict__ x, const float* __restrict__ emb,
    unsigned int* __restrict__ xeA) {
  int bid = blockIdx.x;          // t*32 + b
  int t = bid >> 5, b = bid & 31;
  int row = x[b * Tdim + t];     // x is [B, T]
  const float* src = emb + (size_t)row * Edim;
  unsigned int* dst = xeA + (size_t)t * (Bdim * Edim / 2);
  for (int k = threadIdx.x; k < Edim; k += 256) {
    int dw, half;
    a_slot(b, k, dw, half);
    ((unsigned short*)dst)[dw * 2 + half] = f2bf(src[k]);
  }
}

// ---------------------------------------------------------------------------
// Pack 4 LSTM weight matrices [4096 x 1024] fp32 -> bf16 WMMA B-layout.
// B-tile (32Kx16N): lanes 0-15 hold K 0..15 (V_v: K=2v,2v+1), lanes 16-31 hold
// K 16..31. dword index = ((ntile*32 + kt)*32 + lane)*8 + v. 2 MiW dwords each.
// ---------------------------------------------------------------------------
__global__ void __launch_bounds__(256) weight_pack_kernel(
    const float* __restrict__ W0, const float* __restrict__ W1,
    const float* __restrict__ W2, const float* __restrict__ W3,
    unsigned int* __restrict__ dst) {
  unsigned int gid = blockIdx.x * 256u + threadIdx.x;   // < 4 * 2097152
  unsigned int wsel = gid >> 21;
  unsigned int rem  = gid & 0x1FFFFFu;
  const float* W = (wsel == 0) ? W0 : (wsel == 1) ? W1 : (wsel == 2) ? W2 : W3;
  unsigned int ntile = rem >> 13;           // 256 n-tiles (N = 4096)
  unsigned int kt    = (rem >> 8) & 31;     // 32 k-tiles (K = 1024)
  unsigned int lane  = (rem >> 3) & 31;
  unsigned int v     = rem & 7;
  unsigned int n = ntile * 16 + (lane & 15);
  unsigned int k = kt * 32 + ((lane >> 4) & 1) * 16 + 2 * v;
  const float* p = W + (size_t)n * 1024 + k;  // torch Linear W: [out,in], B=W^T
  dst[gid] = (unsigned int)f2bf(p[0]) | ((unsigned int)f2bf(p[1]) << 16);
}

// ---------------------------------------------------------------------------
// One LSTM cell step (one layer): gates = xA*WiB + hA*WhB + bi + bh, then
// elementwise cell. Grid = 64 blocks (16 h-cols each); block = 8 wave32,
// wave (mt, gate) owns one 16x16 C tile; K-loop = 32+32 WMMAs.
// ---------------------------------------------------------------------------
__global__ void __launch_bounds__(256) lstm_step_kernel(
    const unsigned int* __restrict__ xA,   // packed bf16 A [2][32][32][8]
    const unsigned int* __restrict__ hA,   // packed bf16 A (prev h)
    const unsigned int* __restrict__ WiB,  // packed bf16 B
    const unsigned int* __restrict__ WhB,  // packed bf16 B
    const float* __restrict__ bi, const float* __restrict__ bh,
    float* __restrict__ c,                 // fp32 [32][1024], in-place
    unsigned int* __restrict__ hOutA) {    // packed bf16 A (next h)
  __shared__ float gsm[4][32][16];         // [gate][m][col], 8 KB
  const int tid = threadIdx.x;
  const int lane = tid & 31, w = tid >> 5;
  const int mt = w & 1, gidx = w >> 1;     // gate 0..3 = i,f,g,o
  const int bx = blockIdx.x;               // h-column tile (16 cols)
  const int ntile = gidx * 64 + bx;        // column tile in [4096]

  v8f acc = {};
  const unsigned int* ap = xA + (size_t)(mt * 32) * 256 + lane * 8;
  const unsigned int* bp = WiB + (size_t)(ntile * 32) * 256 + lane * 8;
#pragma unroll 4
  for (int kt = 0; kt < 32; ++kt) {
    v8u a = *(const v8u*)(ap + kt * 256);
    v8u b = *(const v8u*)(bp + kt * 256);
    acc = wmma_bf16(a, b, acc);
  }
  ap = hA + (size_t)(mt * 32) * 256 + lane * 8;
  bp = WhB + (size_t)(ntile * 32) * 256 + lane * 8;
#pragma unroll 4
  for (int kt = 0; kt < 32; ++kt) {
    v8u a = *(const v8u*)(ap + kt * 256);
    v8u b = *(const v8u*)(bp + kt * 256);
    acc = wmma_bf16(a, b, acc);
  }

  // C layout: VGPR r -> M = r + 8*(lane>=16), N = lane&15
  {
    const int col = lane & 15;
    const int rbase = (lane >> 4) * 8;
#pragma unroll
    for (int r = 0; r < 8; ++r)
      gsm[gidx][mt * 16 + rbase + r][col] = acc[r];
  }
  __syncthreads();

  // fused elementwise cell: 32 rows x 16 cols, 2 per thread
  for (int e = tid; e < 512; e += 256) {
    int m = e >> 4, cc = e & 15;
    int hcol = bx * 16 + cc;
    float gi = gsm[0][m][cc] + bi[hcol]        + bh[hcol];
    float gf = gsm[1][m][cc] + bi[1024 + hcol] + bh[1024 + hcol];
    float gg = gsm[2][m][cc] + bi[2048 + hcol] + bh[2048 + hcol];
    float go = gsm[3][m][cc] + bi[3072 + hcol] + bh[3072 + hcol];
    gi = sigmoidf_fast(gi);
    gf = sigmoidf_fast(gf);
    gg = tanhf(gg);
    go = sigmoidf_fast(go);
    float cv = gf * c[m * 1024 + hcol] + gi * gg;   // c is column-local: safe
    c[m * 1024 + hcol] = cv;
    float hv = go * tanhf(cv);
    int dw, half;
    a_slot(m, hcol, dw, half);
    ((unsigned short*)hOutA)[dw * 2 + half] = f2bf(hv);
  }
}

// ---------------------------------------------------------------------------
// Final FC: out[32][32000] = h1 @ fc_w^T + fc_b. HBM-bound on fc_w (128 MB);
// convert fc_w tiles to bf16 B-layout on the fly, WMMA accumulate in f32.
// ---------------------------------------------------------------------------
__global__ void __launch_bounds__(256) fc_kernel(
    const unsigned int* __restrict__ hA, const float* __restrict__ fc_w,
    const float* __restrict__ fc_b, float* __restrict__ out) {
  const int tid = threadIdx.x, lane = tid & 31, w = tid >> 5;
  const int mt = w & 1, nsub = w >> 1;
  const int ntile = blockIdx.x * 4 + nsub;   // 2000 n-tiles total
  const int n0 = ntile * 16;
  const int nn = n0 + (lane & 15);
  const int khalf = ((lane >> 4) & 1) * 16;

  v8f acc = {};
  const unsigned int* ap = hA + (size_t)(mt * 32) * 256 + lane * 8;
  for (int kt = 0; kt < 32; ++kt) {
    v8u a = *(const v8u*)(ap + kt * 256);
    const float2* wp =
        (const float2*)(fc_w + (size_t)nn * 1024 + kt * 32 + khalf);
    v8u b;
#pragma unroll
    for (int v = 0; v < 8; ++v) {
      float2 w2 = wp[v];
      b[v] = (unsigned int)f2bf(w2.x) | ((unsigned int)f2bf(w2.y) << 16);
    }
    acc = wmma_bf16(a, b, acc);
  }
  const int col = lane & 15;
  const int n = n0 + col;
  const float bias = fc_b[n];
  const int rbase = (lane >> 4) * 8;
#pragma unroll
  for (int r = 0; r < 8; ++r) {
    int m = mt * 16 + rbase + r;
    out[(size_t)m * OUTD + n] = acc[r] + bias;
  }
}

// ---------------------------------------------------------------------------
extern "C" void kernel_launch(void* const* d_in, const int* in_sizes, int n_in,
                              void* d_out, int out_size, void* d_ws,
                              size_t ws_size, hipStream_t stream) {
  (void)in_sizes; (void)n_in; (void)out_size; (void)ws_size;
  const int*   x   = (const int*)  d_in[0];
  const float* emb = (const float*)d_in[1];
  const float* Wi0 = (const float*)d_in[2];
  const float* bi0 = (const float*)d_in[3];
  const float* Wh0 = (const float*)d_in[4];
  const float* bh0 = (const float*)d_in[5];
  const float* Wi1 = (const float*)d_in[6];
  const float* bi1 = (const float*)d_in[7];
  const float* Wh1 = (const float*)d_in[8];
  const float* bh1 = (const float*)d_in[9];
  const float* fcw = (const float*)d_in[10];
  const float* fcb = (const float*)d_in[11];

  // workspace layout (~64.6 MB)
  char* ws = (char*)d_ws;
  size_t off = 0;
  unsigned int* xeA = (unsigned int*)(ws + off);
  off += (size_t)Tdim * Bdim * Edim * 2;               // 32 MB packed bf16 xe
  unsigned int* Wp = (unsigned int*)(ws + off);
  off += 4ull * 4096 * 1024 * 2;                       // 32 MB packed bf16 W
  char* stateBase = ws + off;
  unsigned int* h0A[2] = {(unsigned int*)(ws + off), (unsigned int*)(ws + off + 65536)};
  off += 2 * 65536;                                    // h0 ping-pong (bf16)
  unsigned int* h1A[2] = {(unsigned int*)(ws + off), (unsigned int*)(ws + off + 65536)};
  off += 2 * 65536;                                    // h1 ping-pong (bf16)
  float* c0 = (float*)(ws + off); off += (size_t)Bdim * Hdim * 4;
  float* c1 = (float*)(ws + off); off += (size_t)Bdim * Hdim * 4;
  const size_t stateBytes = 4 * 65536 + 2 * (size_t)Bdim * Hdim * 4;

  hipMemsetAsync(stateBase, 0, stateBytes, stream);    // h = c = 0

  embed_pack_kernel<<<Tdim * Bdim, 256, 0, stream>>>(x, emb, xeA);
  weight_pack_kernel<<<(4 * 2097152) / 256, 256, 0, stream>>>(Wi0, Wh0, Wi1,
                                                              Wh1, Wp);
  unsigned int* WiB0 = Wp;
  unsigned int* WhB0 = Wp + 2097152;
  unsigned int* WiB1 = Wp + 2 * 2097152;
  unsigned int* WhB1 = Wp + 3 * 2097152;

  for (int t = 0; t < Tdim; ++t) {
    const unsigned int* xt = xeA + (size_t)t * (Bdim * Edim / 2);
    lstm_step_kernel<<<Hdim / 16, 256, 0, stream>>>(
        xt, h0A[t & 1], WiB0, WhB0, bi0, bh0, c0, h0A[(t + 1) & 1]);
    lstm_step_kernel<<<Hdim / 16, 256, 0, stream>>>(
        h0A[(t + 1) & 1], h1A[t & 1], WiB1, WhB1, bi1, bh1, c1,
        h1A[(t + 1) & 1]);
  }
  // after t=511, latest h1 is in buffer index (512)&1 == 0
  fc_kernel<<<OUTD / 64, 256, 0, stream>>>(h1A[0], fcw, fcb, (float*)d_out);
}
